// standard_GRU_10170482557696
// MI455X (gfx1250) — matrix operand order repaired and checked by reading, not compiled, not run on hardware
//
#include <hip/hip_runtime.h>
#include <hip/hip_bf16.h>
#include <math.h>

// ---------------- problem constants ----------------
#define GRU_B 256
#define GRU_T 2048
#define GRU_I 128
#define GRU_H 256
// LDS row pitches (elements) chosen so row stride % 64 dwords is small+nonzero
#define XP 136   // 128 + 8 bf16  -> 272B = 68 dwords, %64 = 4
#define HP 264   // 256 + 8 bf16  -> 528B = 132 dwords, %64 = 4

typedef __attribute__((ext_vector_type(16))) __bf16        v16bf;
typedef __attribute__((ext_vector_type(8)))  float         v8f;
typedef __attribute__((ext_vector_type(4)))  float         v4f;
typedef __attribute__((ext_vector_type(4)))  unsigned int  v4u;
typedef __attribute__((ext_vector_type(2)))  unsigned int  v2u;

union Frag { v16bf bf; v4u q[2]; unsigned short us[16]; };

// Explicit global-address-space pointer: guarantees global_load_b128 (not flat).
// v4u is a builtin vector type, so AS-qualified loads are plain loads (no
// implicit member functions -> compiles on both host and device passes).
typedef const v4u __attribute__((address_space(1))) *gv4up;

__device__ __forceinline__ unsigned short f2bf(float f) {
  unsigned int u = __float_as_uint(f);
  u += 0x7FFFu + ((u >> 16) & 1u);   // round-to-nearest-even
  return (unsigned short)(u >> 16);
}

__device__ __forceinline__ unsigned int pack2bf(float a, float b) {
  return (unsigned int)f2bf(a) | ((unsigned int)f2bf(b) << 16);
}

__device__ __forceinline__ float sigmoid_f(float x) {
  return 1.0f / (1.0f + __expf(-x));
}

__device__ __forceinline__ v8f wmma_bf16(const Frag& a, const Frag& b, v8f c) {
  return __builtin_amdgcn_wmma_f32_16x16x32_bf16(
      /*neg_a=*/false, a.bf, /*neg_b=*/false, b.bf,
      /*c_mod=*/(short)0, c, /*reuse_a=*/false, /*reuse_b=*/false);
}

// A fragment (16x32 bf16) from LDS: lane L(0..15) row M=L,
// K = kc*32 + hi*8 .. +7 and +16..+23
__device__ __forceinline__ void loadA(Frag& f, const unsigned short* base) {
  f.q[0] = *(const v4u*)(base);
  f.q[1] = *(const v4u*)(base + 16);
}
// B fragment from global (packed contiguously, 16 bf16 per lane)
__device__ __forceinline__ void loadB(Frag& f, const unsigned short* base) {
  gv4up p = (gv4up)base;            // addrspacecast generic -> global
  f.q[0] = p[0];
  f.q[1] = p[1];
}

// ---------------------------------------------------------------------------
// Pack input-side weights W[g] (H x I, f32, row-major) into per-lane bf16
// B-fragment layout: dst[((g*16 + n)*4 + kc)*32 + l][16 elems]
//   lane l: N = l&15 (col n*16+N of output), khalf = l>>4
//   elems j: K = kc*32 + khalf*8 + j (j<8), kc*32 + 16 + khalf*8 + (j-8) (j>=8)  ->  W[row=n*16+N][K]
// ---------------------------------------------------------------------------
__global__ void pack_wx(const float* __restrict__ wrx, const float* __restrict__ wzx,
                        const float* __restrict__ whx, unsigned short* __restrict__ dst) {
  int idx = blockIdx.x * 256 + threadIdx.x;           // 3 * 16*4*32*16 = 98304
  if (idx >= 3 * 16 * 4 * 32 * 16) return;
  int j  = idx & 15;
  int l  = (idx >> 4) & 31;
  int kc = (idx >> 9) & 3;
  int n  = (idx >> 11) & 15;
  int g  = idx >> 15;
  const float* w = (g == 0) ? wrx : ((g == 1) ? wzx : whx);
  int row = n * 16 + (l & 15);
  int col = kc * 32 + (j < 8 ? (l >> 4) * 8 + j : 16 + (l >> 4) * 8 + (j - 8));
  dst[idx] = f2bf(w[row * GRU_I + col]);
}

__global__ void pack_wh(const float* __restrict__ wrh, const float* __restrict__ wzh,
                        const float* __restrict__ whh, unsigned short* __restrict__ dst) {
  int idx = blockIdx.x * 256 + threadIdx.x;           // 3 * 16*8*32*16 = 196608
  if (idx >= 3 * 16 * 8 * 32 * 16) return;
  int j  = idx & 15;
  int l  = (idx >> 4) & 31;
  int kc = (idx >> 9) & 7;
  int n  = (idx >> 12) & 15;
  int g  = idx >> 16;
  const float* w = (g == 0) ? wrh : ((g == 1) ? wzh : whh);
  int row = n * 16 + (l & 15);
  int col = kc * 32 + (j < 8 ? (l >> 4) * 8 + j : 16 + (l >> 4) * 8 + (j - 8));
  dst[idx] = f2bf(w[row * GRU_H + col]);
}

// ---------------------------------------------------------------------------
// Persistent GRU scan. grid = B/16 workgroups, 512 threads = 16 waves.
// Workgroup owns batch rows [b0, b0+16); wave w owns hidden columns
// [w*16, w*16+16). h lives in LDS (f32 master + bf16 fragment copy).
// Weights stay in L2 and are re-fetched each step; a per-iteration opaque
// zero-offset stops the compiler from hoisting ~288 VGPRs of fragments
// (which would spill), while keeping pointer provenance (global, not flat).
// ---------------------------------------------------------------------------
__global__ void __launch_bounds__(512, 1)
gru_scan(const float* __restrict__ x,            // [B,T,I] f32
         const unsigned short* __restrict__ wxP, // packed [3][16][4][32][16] bf16
         const unsigned short* __restrict__ whP, // packed [3][16][8][32][16] bf16
         const float* __restrict__ br, const float* __restrict__ bz,
         const float* __restrict__ bh,
         float* __restrict__ out) {              // [B,T,H] f32
  __shared__ unsigned short x_bf[16 * XP];
  __shared__ unsigned short h_bf[16 * HP];
  __shared__ unsigned short rh_bf[16 * HP];
  __shared__ float h_f32[16 * HP];

  const int tid  = threadIdx.x;
  const int wave = tid >> 5;      // 0..15 -> column tile
  const int lane = tid & 31;
  const int lN   = lane & 15;     // N within 16-col tile / A-row M
  const int hi   = lane >> 4;     // K-half selector for A fragments
  const int b0   = blockIdx.x * 16;
  const int n0   = wave * 16;

  for (int i = tid; i < 16 * HP; i += 512) { h_bf[i] = 0; h_f32[i] = 0.0f; }

  const float biasR = br[n0 + lN];
  const float biasZ = bz[n0 + lN];
  const float biasH = bh[n0 + lN];

  // x staging assignment: each of 16 rows handled by 32 threads * float4
  const int srow = tid >> 5;
  const int sc4  = (tid & 31) * 4;

  __syncthreads();

  for (int t = 0; t < GRU_T; ++t) {
    // Anti-LICM: opaque zero offset makes weight addresses loop-variant while
    // preserving GEP provenance from the kernel args (=> global_load path).
    size_t z = 0;
    asm volatile("" : "+s"(z));
    const unsigned short* wxp = wxP + z;
    const unsigned short* whp = whP + z;

    // ---- stage x tile [16 x 128] f32 -> bf16 LDS, prefetch next step ----
    {
      const v4f v = __builtin_nontemporal_load(
          (const v4f*)&x[(size_t)((b0 + srow) * GRU_T + t) * GRU_I + sc4]);
      v2u p; p.x = pack2bf(v.x, v.y); p.y = pack2bf(v.z, v.w);
      *(v2u*)&x_bf[srow * XP + sc4] = p;
      if (t + 1 < GRU_T)
        __builtin_prefetch(&x[(size_t)((b0 + srow) * GRU_T + (t + 1)) * GRU_I + sc4], 0, 0);
    }
    __syncthreads();

    v8f accR = {}, accZ = {}, accH = {};

    // ---- input projections: K = 128, 4 chunks, 3 gates -> 12 WMMAs ----
#pragma unroll
    for (int kc = 0; kc < 4; ++kc) {
      Frag fa; loadA(fa, &x_bf[lN * XP + kc * 32 + hi * 8]);
      const size_t wb = ((size_t)(wave * 4 + kc) * 32 + lane) * 16;
      Frag fr, fz, fh;
      loadB(fr, wxp + 0 * 32768 + wb);
      loadB(fz, wxp + 1 * 32768 + wb);
      loadB(fh, wxp + 2 * 32768 + wb);
      accR = wmma_bf16(fa, fr, accR);
      accZ = wmma_bf16(fa, fz, accZ);
      accH = wmma_bf16(fa, fh, accH);
    }

    // ---- recurrent r,z: K = 256, 8 chunks, 2 gates -> 16 WMMAs ----
#pragma unroll
    for (int kc = 0; kc < 8; ++kc) {
      Frag fa; loadA(fa, &h_bf[lN * HP + kc * 32 + hi * 8]);
      const size_t wb = ((size_t)(wave * 8 + kc) * 32 + lane) * 16;
      Frag fr, fz;
      loadB(fr, whp + 0 * 65536 + wb);
      loadB(fz, whp + 1 * 65536 + wb);
      accR = wmma_bf16(fa, fr, accR);
      accZ = wmma_bf16(fa, fz, accZ);
    }

    // ---- r = sigmoid(.), write r*h (bf16) for the h~ GEMM ----
    float hold[8];
#pragma unroll
    for (int m = 0; m < 8; ++m) {
      const int M = m + 8 * hi;                  // C-layout row for vgpr m
      hold[m] = h_f32[M * HP + n0 + lN];
      const float rv = sigmoid_f(accR[m] + biasR);
      rh_bf[M * HP + n0 + lN] = f2bf(rv * hold[m]);
    }
    __syncthreads();

    // ---- recurrent h~: K = 256, 8 chunks -> 8 WMMAs ----
#pragma unroll
    for (int kc = 0; kc < 8; ++kc) {
      Frag fa; loadA(fa, &rh_bf[lN * HP + kc * 32 + hi * 8]);
      const size_t wb = ((size_t)(wave * 8 + kc) * 32 + lane) * 16;
      Frag fh; loadB(fh, whp + 2 * 65536 + wb);
      accH = wmma_bf16(fa, fh, accH);
    }

    // ---- blend, commit h, stream to output (non-temporal) ----
#pragma unroll
    for (int m = 0; m < 8; ++m) {
      const int M = m + 8 * hi;
      const float zv = sigmoid_f(accZ[m] + biasZ);
      const float ht = tanhf(accH[m] + biasH);
      const float hn = (1.0f - zv) * hold[m] + zv * ht;
      h_f32[M * HP + n0 + lN] = hn;
      h_bf [M * HP + n0 + lN] = f2bf(hn);
      __builtin_nontemporal_store(
          hn, &out[(size_t)((b0 + M) * GRU_T + t) * GRU_H + n0 + lN]);
    }
    // next iteration's staging barrier also fences these LDS writes
  }
}

// ---------------------------------------------------------------------------
extern "C" void kernel_launch(void* const* d_in, const int* in_sizes, int n_in,
                              void* d_out, int out_size, void* d_ws, size_t ws_size,
                              hipStream_t stream) {
  (void)in_sizes; (void)n_in; (void)out_size; (void)ws_size;
  const float* x    = (const float*)d_in[0];
  const float* w_rx = (const float*)d_in[1];
  const float* w_rh = (const float*)d_in[2];
  const float* b_r  = (const float*)d_in[3];
  const float* w_zx = (const float*)d_in[4];
  const float* w_zh = (const float*)d_in[5];
  const float* b_z  = (const float*)d_in[6];
  const float* w_hx = (const float*)d_in[7];
  const float* w_hh = (const float*)d_in[8];
  const float* b_h  = (const float*)d_in[9];
  float* out = (float*)d_out;

  unsigned short* wxP = (unsigned short*)d_ws;       // 3*32768 bf16 = 192 KB
  unsigned short* whP = wxP + 3 * 32768;             // 3*65536 bf16 = 384 KB

  pack_wx<<<(3 * 32768) / 256, 256, 0, stream>>>(w_rx, w_zx, w_hx, wxP);
  pack_wh<<<(3 * 65536) / 256, 256, 0, stream>>>(w_rh, w_zh, w_hh, whP);
  gru_scan<<<GRU_B / 16, 512, 0, stream>>>(x, wxP, whP, b_r, b_z, b_h, out);
}
